// MHA_22436909154704
// MI455X (gfx1250) — matrix operand order repaired and checked by reading, not compiled, not run on hardware
//
#include <hip/hip_runtime.h>
#include <hip/hip_bf16.h>

// ---------------------------------------------------------------------------
// MHA forward for MI455X (gfx1250), wave32, f16 WMMA w/ f32 accumulation.
//   B=2, L=2048, D=1024, H=16, hd=64
// All GEMM hot loops are pure-f16 b128 fragment loads + v_wmma; fp32->f16
// conversion is hoisted into one streaming pre-pass.
// ---------------------------------------------------------------------------

typedef __attribute__((ext_vector_type(16))) _Float16 v16h;
typedef __attribute__((ext_vector_type(8)))  float    v8f;

#define BATCH   2
#define SEQ     2048
#define DMODEL  1024
#define NHEAD   16
#define HDIM    64
#define MROWS   (BATCH * SEQ)          /* 4096 */
#define NQKV    (3 * DMODEL)           /* 3072 */

union F16Frag {
    v16h v;
    uint4 q[2];
};
union F16x8 {
    uint4 q;
    _Float16 h[8];
};

// ---- fragment loaders (contiguous k = 16h+s order, same for A and B) -------------
// A (16x32 MxK): lane m=lane&15, half h=lane>>4 holds K = 16h..16h+15
// contiguously -> two contiguous 16B loads from a row-major K row.
__device__ inline v16h load_a_f16(const _Float16* rowk, int h) {
    F16Frag u;
    u.q[0] = *(const uint4*)(rowk + 16 * h);
    u.q[1] = *(const uint4*)(rowk + 16 * h + 8);
    return u.v;
}
// B (32x16 KxN): lane n=lane&15, half h holds K = 16h..16h+15 contiguously.
// Caller passes pointer already offset by +16h along K.
__device__ inline v16h load_b_f16(const _Float16* p) {
    F16Frag u;
    u.q[0] = *(const uint4*)(p);
    u.q[1] = *(const uint4*)(p + 8);
    return u.v;
}

__device__ inline v8f wmma16(v16h a, v16h b, v8f c) {
    return __builtin_amdgcn_wmma_f32_16x16x32_f16(
        /*neg_a=*/false, a, /*neg_b=*/false, b,
        /*c_mod=*/(short)0, c, /*reuse_a=*/false, /*reuse_b=*/false);
}

__device__ inline float rowmax16(float v) {
#pragma unroll
    for (int m = 1; m < 16; m <<= 1) v = fmaxf(v, __shfl_xor(v, m, 32));
    return v;
}
__device__ inline float rowsum16(float v) {
#pragma unroll
    for (int m = 1; m < 16; m <<= 1) v += __shfl_xor(v, m, 32);
    return v;
}

// ---------------------------------------------------------------------------
// Kernel 0: streaming fp32 -> f16 conversion (8 elements / thread, b128 I/O).
// ---------------------------------------------------------------------------
__global__ __launch_bounds__(256)
void cvt_f32_to_f16(const float* __restrict__ src, _Float16* __restrict__ dst,
                    int n8) {
    const int i = blockIdx.x * blockDim.x + threadIdx.x;
    if (i >= n8) return;
    const float4* s = (const float4*)src + (size_t)i * 2;
    const float4 a = s[0];
    const float4 b = s[1];
    F16x8 u;
    u.h[0] = (_Float16)a.x; u.h[1] = (_Float16)a.y;
    u.h[2] = (_Float16)a.z; u.h[3] = (_Float16)a.w;
    u.h[4] = (_Float16)b.x; u.h[5] = (_Float16)b.y;
    u.h[6] = (_Float16)b.z; u.h[7] = (_Float16)b.w;
    *((uint4*)dst + i) = u.q;
}

// ---------------------------------------------------------------------------
// Kernel 1: fused QKV projection.  qkv = x_h @ Wqkv_h^T + b_qkv (all-f16 frags)
// One wave -> 16x64 output tile. Scatter Q,K -> [B,H,L,hd]; V -> [B,H,hd,L].
// ---------------------------------------------------------------------------
__global__ __launch_bounds__(256)
void mha_qkv_proj(const _Float16* __restrict__ xh, const _Float16* __restrict__ Wh,
                  const float* __restrict__ bqkv,
                  _Float16* __restrict__ Qh, _Float16* __restrict__ Kh,
                  _Float16* __restrict__ Vt) {
    const int wid    = (blockIdx.x * blockDim.x + threadIdx.x) >> 5;
    const int lane   = threadIdx.x & 31;
    const int lane15 = lane & 15;
    const int hl     = lane >> 4;

    const int nQuads = NQKV / 64;              // 48
    const int mTile  = wid / nQuads;           // 0..255
    const int nQuad  = wid % nQuads;

    v8f acc[4] = {};
    const _Float16* xrow = xh + (size_t)(mTile * 16 + lane15) * DMODEL;

    for (int k0 = 0; k0 < DMODEL; k0 += 32) {
        v16h a = load_a_f16(xrow + k0, hl);
#pragma unroll
        for (int t = 0; t < 4; ++t) {
            const int n = nQuad * 64 + t * 16 + lane15;
            v16h b = load_b_f16(Wh + (size_t)n * DMODEL + k0 + 16 * hl);
            acc[t] = wmma16(a, b, acc[t]);
        }
    }

#pragma unroll
    for (int t = 0; t < 4; ++t) {
        const int n    = nQuad * 64 + t * 16 + lane15;
        const float bias = bqkv[n];
        const int seg  = n >> 10;              // 0:Q 1:K 2:V (64-aligned tiles)
        const int nn   = n & 1023;
        const int head = nn >> 6;
        const int d    = nn & 63;
#pragma unroll
        for (int r = 0; r < 8; ++r) {
            const int m  = mTile * 16 + r + 8 * hl;
            const int bb = m >> 11;
            const int l  = m & 2047;
            const _Float16 val = (_Float16)(acc[t][r] + bias);
            if (seg == 0)
                Qh[(((size_t)(bb * NHEAD + head)) * SEQ + l) * HDIM + d] = val;
            else if (seg == 1)
                Kh[(((size_t)(bb * NHEAD + head)) * SEQ + l) * HDIM + d] = val;
            else
                Vt[(((size_t)(bb * NHEAD + head)) * HDIM + d) * SEQ + l] = val;
        }
    }
}

// ---------------------------------------------------------------------------
// Kernel 2: flash attention. One wave owns a 16-query tile of one (b,h) and
// streams keys in blocks of 32: 4 WMMAs for S = Q K^T, online softmax,
// P goes through LDS (C-layout -> A-layout), 4 WMMAs for O += P V.
// ---------------------------------------------------------------------------
__global__ __launch_bounds__(256)
void mha_flash_attn(const _Float16* __restrict__ Qh, const _Float16* __restrict__ Kh,
                    const _Float16* __restrict__ Vt, _Float16* __restrict__ AttnOut) {
    __shared__ _Float16 Plds[8][16][32];       // per-wave 16x32 P tile (8 KB)

    const int wid    = (blockIdx.x * blockDim.x + threadIdx.x) >> 5;
    const int lane   = threadIdx.x & 31;
    const int lane15 = lane & 15;
    const int hl     = lane >> 4;
    const int w      = threadIdx.x >> 5;

    const int qt   = wid & 127;                // query tile 0..127
    const int head = (wid >> 7) & 15;
    const int bb   = wid >> 11;                // batch

    const _Float16* Qbase = Qh + ((size_t)(bb * NHEAD + head)) * SEQ * HDIM;
    const _Float16* Kbase = Kh + ((size_t)(bb * NHEAD + head)) * SEQ * HDIM;
    const _Float16* Vbase = Vt + ((size_t)(bb * NHEAD + head)) * HDIM * SEQ;

    // Q fragments for d = 0..31 and 32..63, reused for all key blocks.
    v16h qf[2];
#pragma unroll
    for (int ds = 0; ds < 2; ++ds)
        qf[ds] = load_a_f16(Qbase + (size_t)(qt * 16 + lane15) * HDIM + ds * 32, hl);

    float mrun[8], lrun[8];
#pragma unroll
    for (int r = 0; r < 8; ++r) { mrun[r] = -1e30f; lrun[r] = 0.0f; }
    v8f o[4] = {};

    const float SCALE = 0.125f;                // 1/sqrt(64)

    for (int kb = 0; kb < SEQ; kb += 32) {
        // ---- S tile: 16 queries x 32 keys --------------------------------
        v8f s[2] = {};
#pragma unroll
        for (int sub = 0; sub < 2; ++sub) {
            const int key = kb + sub * 16 + lane15;
#pragma unroll
            for (int ds = 0; ds < 2; ++ds) {
                v16h kf = load_b_f16(Kbase + (size_t)key * HDIM + ds * 32 + 16 * hl);
                s[sub] = wmma16(qf[ds], kf, s[sub]);
            }
        }
        // ---- online softmax ----------------------------------------------
#pragma unroll
        for (int r = 0; r < 8; ++r) {
            const float s0 = s[0][r] * SCALE;
            const float s1 = s[1][r] * SCALE;
            const float vm   = rowmax16(fmaxf(s0, s1));
            const float mnew = fmaxf(mrun[r], vm);
            const float alpha = __expf(mrun[r] - mnew);
            const float p0 = __expf(s0 - mnew);
            const float p1 = __expf(s1 - mnew);
            lrun[r] = lrun[r] * alpha + rowsum16(p0 + p1);
            mrun[r] = mnew;
#pragma unroll
            for (int t = 0; t < 4; ++t) o[t][r] *= alpha;
            Plds[w][r + 8 * hl][lane15]      = (_Float16)p0;
            Plds[w][r + 8 * hl][16 + lane15] = (_Float16)p1;
        }
        // ---- P (A-layout) from LDS, then O += P V ------------------------
        F16Frag pf;
        pf.q[0] = *(const uint4*)&Plds[w][lane15][16 * hl];
        pf.q[1] = *(const uint4*)&Plds[w][lane15][16 * hl + 8];
#pragma unroll
        for (int t = 0; t < 4; ++t) {
            const int d = t * 16 + lane15;
            v16h vf = load_b_f16(Vbase + (size_t)d * SEQ + kb + 16 * hl);
            o[t] = wmma16(pf.v, vf, o[t]);
        }
    }

    // ---- normalize and store attention output as f16 [B*L, D] -----------
#pragma unroll
    for (int r = 0; r < 8; ++r) {
        const float inv = 1.0f / lrun[r];
        const int m = qt * 16 + r + 8 * hl;
#pragma unroll
        for (int t = 0; t < 4; ++t) {
            const int col = head * HDIM + t * 16 + lane15;
            AttnOut[((size_t)(bb * SEQ + m)) * DMODEL + col] = (_Float16)(o[t][r] * inv);
        }
    }
}

// ---------------------------------------------------------------------------
// Kernel 3: output projection. out = AttnOut @ Wout_h^T + b_out -> f32
// ---------------------------------------------------------------------------
__global__ __launch_bounds__(256)
void mha_out_proj(const _Float16* __restrict__ A, const _Float16* __restrict__ Wh,
                  const float* __restrict__ bout, float* __restrict__ out) {
    const int wid    = (blockIdx.x * blockDim.x + threadIdx.x) >> 5;
    const int lane   = threadIdx.x & 31;
    const int lane15 = lane & 15;
    const int hl     = lane >> 4;

    const int nQuads = DMODEL / 64;            // 16
    const int mTile  = wid / nQuads;
    const int nQuad  = wid % nQuads;

    v8f acc[4] = {};
    const _Float16* arow = A + (size_t)(mTile * 16 + lane15) * DMODEL;

    for (int k0 = 0; k0 < DMODEL; k0 += 32) {
        v16h a = load_a_f16(arow + k0, hl);
#pragma unroll
        for (int t = 0; t < 4; ++t) {
            const int n = nQuad * 64 + t * 16 + lane15;
            v16h b = load_b_f16(Wh + (size_t)n * DMODEL + k0 + 16 * hl);
            acc[t] = wmma16(a, b, acc[t]);
        }
    }

#pragma unroll
    for (int t = 0; t < 4; ++t) {
        const int n = nQuad * 64 + t * 16 + lane15;
        const float bias = bout[n];
#pragma unroll
        for (int r = 0; r < 8; ++r) {
            const int m = mTile * 16 + r + 8 * hl;
            out[(size_t)m * DMODEL + n] = acc[t][r] + bias;
        }
    }
}

// ---------------------------------------------------------------------------
extern "C" void kernel_launch(void* const* d_in, const int* in_sizes, int n_in,
                              void* d_out, int out_size, void* d_ws, size_t ws_size,
                              hipStream_t stream) {
    const float* x    = (const float*)d_in[0];   // [2,2048,1024]
    const float* Wqkv = (const float*)d_in[1];   // [3072,1024]
    const float* bqkv = (const float*)d_in[2];   // [3072]
    const float* Wout = (const float*)d_in[3];   // [1024,1024]
    const float* bout = (const float*)d_in[4];   // [1024]
    float* out = (float*)d_out;                  // [2,2048,1024]

    // Workspace layout (f16 elements), 38 MB total with aliasing:
    //   [0,4Mi)   Qh        [4Mi,8Mi)  Kh       [8Mi,12Mi) Vt
    //   [12Mi,16Mi) x_h  -- dead after QKV GEMM, reused as AttnOut
    //   [16Mi,19Mi) Wqkv_h -- dead after QKV GEMM, reused as Wout_h
    const size_t HALF = (size_t)BATCH * NHEAD * SEQ * HDIM;   // 4 Mi
    _Float16* Qh = (_Float16*)d_ws;
    _Float16* Kh = Qh + HALF;
    _Float16* Vt = Kh + HALF;
    _Float16* XA = Vt + HALF;                    // x_h, later AttnOut
    _Float16* Wh = XA + HALF;                    // Wqkv_h, later Wout_h

    // --- f32 -> f16 pre-conversion (streaming, one pass each) ------------
    cvt_f32_to_f16<<<(MROWS * DMODEL / 8) / 256, 256, 0, stream>>>(x, XA, MROWS * DMODEL / 8);
    cvt_f32_to_f16<<<(NQKV * DMODEL / 8) / 256, 256, 0, stream>>>(Wqkv, Wh, NQKV * DMODEL / 8);

    // --- QKV projection: 256 m-tiles x 48 n-quads = 12288 waves ----------
    mha_qkv_proj<<<1536, 256, 0, stream>>>(XA, Wh, bqkv, Qh, Kh, Vt);

    // --- convert W_out into the (now dead) Wqkv_h slot -------------------
    cvt_f32_to_f16<<<(DMODEL * DMODEL / 8) / 256, 256, 0, stream>>>(Wout, Wh, DMODEL * DMODEL / 8);

    // --- flash attention: 2*16*128 = 4096 waves; writes AttnOut over x_h -
    mha_flash_attn<<<512, 256, 0, stream>>>(Qh, Kh, Vt, XA);

    // --- output projection: 256 m-tiles x 16 n-quads = 4096 waves --------
    mha_out_proj<<<512, 256, 0, stream>>>(XA, Wh, bout, out);
}
